// GAT_GraphConv_14465449853242
// MI455X (gfx1250) — matrix operand order repaired and checked by reading, not compiled, not run on hardware
//
#include <hip/hip_runtime.h>
#include <hip/hip_bf16.h>
#include <math.h>

#define N_NODES   50000
#define N_EDGES   400000
#define E_TOT     (N_EDGES + N_NODES)
#define D_FEAT    64
#define H_HEADS   8
#define HD        512          // H * per-head dim = 8 * 64
#define NEG_SLOPE 0.2f
#define LN_EPS    1e-5f

typedef __attribute__((ext_vector_type(16))) __bf16 v16bf;
typedef __attribute__((ext_vector_type(8)))  float  v8f;

// ---------------------------------------------------------------------------
// Float atomic max via sign-split int/uint trick (compiles everywhere).
// ---------------------------------------------------------------------------
__device__ __forceinline__ void atomicMaxF(float* addr, float val) {
    if (val >= 0.0f)
        atomicMax((int*)addr, __float_as_int(val));
    else
        atomicMin((unsigned int*)addr, (unsigned int)__float_as_int(val));
}

// ---------------------------------------------------------------------------
// Pack Wl/Wr (fp32, [3][64][512]) into bf16 WMMA B-fragment order.
// Fragment = (layer, mat, ntile, kstep); within fragment: lane*16 + j, where
//   lane n  (n<16):  col = ntile*16+n,  K = kstep*32 + {0..7, 16..23}[j]
//   lane n+16:       col = ntile*16+n,  K = kstep*32 + {8..15, 24..31}[j]
// ---------------------------------------------------------------------------
__global__ void pack_w_kernel(const float* __restrict__ Wl,
                              const float* __restrict__ Wr,
                              __bf16* __restrict__ wp) {
    int idx = blockIdx.x * 256 + threadIdx.x;
    if (idx >= 3 * 2 * 32 * 2 * 32 * 16) return;
    int j     = idx & 15;
    int lane  = (idx >> 4) & 31;
    int kstep = (idx >> 9) & 1;
    int ntile = (idx >> 10) & 31;
    int mat   = (idx >> 15) & 1;
    int l     = idx >> 16;
    int K = kstep * 32 + ((j & 8) ? 16 : 0) + ((lane & 16) ? 8 : 0) + (j & 7);
    int n = ntile * 16 + (lane & 15);
    const float* W = (mat ? Wr : Wl) + (size_t)l * D_FEAT * HD;
    wp[idx] = (__bf16)W[(size_t)K * HD + n];
}

// ---------------------------------------------------------------------------
// GEMM: one wave computes a 16(M) x 64(N) strip of xl or xr (bf16 out).
// A = x[mtile*16 .. +16, 0..64) fp32 -> bf16 fragments (documented layout):
//   lane<16 : row m0+lane, frag s holds K in {32s+0..7, 32s+16..23}
//   lane>=16: row m0+(lane-16), K in {32s+8..15, 32s+24..31}
// ---------------------------------------------------------------------------
__global__ void gemm_xlxr_kernel(const float* __restrict__ x,
                                 const __bf16* __restrict__ wp,
                                 __bf16* __restrict__ xl,
                                 __bf16* __restrict__ xr,
                                 int layer) {
    const int lane = threadIdx.x & 31;
    const int gw   = blockIdx.x * 8 + (threadIdx.x >> 5);
    const int wsel  = gw & 1;          // 0 -> Wl/xl, 1 -> Wr/xr
    const int nslab = (gw >> 1) & 7;   // 8 slabs of 64 cols
    const int mtile = gw >> 4;         // 3125 row tiles of 16
    if (mtile >= N_NODES / 16) return;

    // ----- A fragments -----
    const int    r    = mtile * 16 + (lane & 15);
    const int    hi   = (lane >> 4) & 1;
    const float* xrow = x + (size_t)r * D_FEAT;
    v16bf afrag[2];
#pragma unroll
    for (int s = 0; s < 2; ++s) {
        const int k0 = s * 32 + hi * 8;
        float4 f0 = *reinterpret_cast<const float4*>(xrow + k0);
        float4 f1 = *reinterpret_cast<const float4*>(xrow + k0 + 4);
        float4 f2 = *reinterpret_cast<const float4*>(xrow + k0 + 16);
        float4 f3 = *reinterpret_cast<const float4*>(xrow + k0 + 20);
        float tmp[16] = {f0.x, f0.y, f0.z, f0.w, f1.x, f1.y, f1.z, f1.w,
                         f2.x, f2.y, f2.z, f2.w, f3.x, f3.y, f3.z, f3.w};
        v16bf a;
#pragma unroll
        for (int j = 0; j < 16; ++j) a[j] = (__bf16)tmp[j];
        afrag[s] = a;
    }

    // ----- B fragments + WMMA accumulation -----
    union BFr { uint4 q[2]; v16bf v; };
    v8f zero = {0.f, 0.f, 0.f, 0.f, 0.f, 0.f, 0.f, 0.f};
    v8f acc[4] = {zero, zero, zero, zero};

#pragma unroll
    for (int t = 0; t < 4; ++t) {
        const int ntile = nslab * 4 + t;
#pragma unroll
        for (int s = 0; s < 2; ++s) {
            const size_t frag = ((size_t)((layer * 2 + wsel) * 32 + ntile)) * 2 + s;
            const uint4* bp = reinterpret_cast<const uint4*>(wp) + frag * 64 + lane * 2;
            BFr bf;
            bf.q[0] = bp[0];
            bf.q[1] = bp[1];
            acc[t] = __builtin_amdgcn_wmma_f32_16x16x32_bf16(
                false, afrag[s], false, bf.v, (short)0, acc[t], false, false);
        }
    }

    // ----- Store D (C/D layout: VGPR v -> M = v + 8*hi, lane&15 -> N) -----
    __bf16* out = wsel ? xr : xl;
    const int colb = nslab * 64 + (lane & 15);
    const int rowb = mtile * 16 + hi * 8;
#pragma unroll
    for (int t = 0; t < 4; ++t)
#pragma unroll
        for (int v = 0; v < 8; ++v)
            out[(size_t)(rowb + v) * HD + colb + t * 16] = (__bf16)acc[t][v];
}

// ---------------------------------------------------------------------------
// Edge logits: one wave per edge. lane covers 16 contiguous channels; 4 lanes
// per head. e_h = att_h . leaky_relu(xl[src]+xr[dst]); atomic max per (dst,h).
// ---------------------------------------------------------------------------
__global__ void edge_logits_kernel(const int* __restrict__ ei,
                                   const __bf16* __restrict__ xl,
                                   const __bf16* __restrict__ xr,
                                   const float* __restrict__ attl,
                                   float* __restrict__ elog,
                                   float* __restrict__ nodemax) {
    const int lane = threadIdx.x & 31;
    const int e    = blockIdx.x * 8 + (threadIdx.x >> 5);
    if (e >= E_TOT) return;
    int src, dst;
    if (e < N_EDGES) { src = ei[e]; dst = ei[N_EDGES + e]; }
    else             { src = dst = e - N_EDGES; }

    const uint4* pl = reinterpret_cast<const uint4*>(xl + (size_t)src * HD + lane * 16);
    const uint4* pr = reinterpret_cast<const uint4*>(xr + (size_t)dst * HD + lane * 16);
    union U8 { uint4 q; __bf16 h[8]; } a0, a1, b0, b1;
    a0.q = pl[0]; a1.q = pl[1];
    b0.q = pr[0]; b1.q = pr[1];

    const float* ap = attl + lane * 16;
    float p = 0.f;
#pragma unroll
    for (int j = 0; j < 8; ++j) {
        float s0 = (float)a0.h[j] + (float)b0.h[j];
        s0 = s0 > 0.f ? s0 : NEG_SLOPE * s0;
        p += s0 * ap[j];
        float s1 = (float)a1.h[j] + (float)b1.h[j];
        s1 = s1 > 0.f ? s1 : NEG_SLOPE * s1;
        p += s1 * ap[8 + j];
    }
    p += __shfl_xor(p, 1, 32);
    p += __shfl_xor(p, 2, 32);
    if ((lane & 3) == 0) {
        const int h = lane >> 2;
        elog[(size_t)e * H_HEADS + h] = p;
        atomicMaxF(&nodemax[dst * H_HEADS + h], p);
    }
}

// ---------------------------------------------------------------------------
// exp(e - max) and segment sum (one thread per (edge, head)).
// ---------------------------------------------------------------------------
__global__ void edge_expsum_kernel(const int* __restrict__ ei,
                                   float* __restrict__ elog,
                                   const float* __restrict__ nodemax,
                                   float* __restrict__ nodesum) {
    int t = blockIdx.x * 256 + threadIdx.x;
    if (t >= E_TOT * H_HEADS) return;
    int e = t >> 3, h = t & 7;
    int dst = (e < N_EDGES) ? ei[N_EDGES + e] : (e - N_EDGES);
    float ex = __expf(elog[t] - nodemax[dst * H_HEADS + h]);
    elog[t] = ex;
    atomicAdd(&nodesum[dst * H_HEADS + h], ex);
}

// ---------------------------------------------------------------------------
// Aggregation: one wave per edge. Head-mean folded in (alpha * 1/8); lanes
// sharing the same channel range across the 8 heads reduce via shfl_xor over
// lane bits 2..4, then 4 lanes issue 16 fp32 atomics each (64/edge).
// ---------------------------------------------------------------------------
__global__ void edge_aggregate_kernel(const int* __restrict__ ei,
                                      const __bf16* __restrict__ xl,
                                      const float* __restrict__ elog,
                                      const float* __restrict__ nodesum,
                                      float* __restrict__ acc64) {
    const int lane = threadIdx.x & 31;
    const int e    = blockIdx.x * 8 + (threadIdx.x >> 5);
    if (e >= E_TOT) return;
    int src, dst;
    if (e < N_EDGES) { src = ei[e]; dst = ei[N_EDGES + e]; }
    else             { src = dst = e - N_EDGES; }

    const int h = lane >> 2;
    const float alpha = elog[(size_t)e * H_HEADS + h] /
                        nodesum[dst * H_HEADS + h] * 0.125f;

    const uint4* pl = reinterpret_cast<const uint4*>(xl + (size_t)src * HD + lane * 16);
    union U8 { uint4 q; __bf16 h[8]; } a0, a1;
    a0.q = pl[0]; a1.q = pl[1];

    float vals[16];
#pragma unroll
    for (int j = 0; j < 8; ++j) {
        vals[j]     = (float)a0.h[j] * alpha;
        vals[8 + j] = (float)a1.h[j] * alpha;
    }
#pragma unroll
    for (int m = 4; m < 32; m <<= 1)
#pragma unroll
        for (int j = 0; j < 16; ++j)
            vals[j] += __shfl_xor(vals[j], m, 32);

    if (lane < 4) {
        float* d = acc64 + (size_t)dst * D_FEAT + lane * 16;
#pragma unroll
        for (int j = 0; j < 16; ++j) atomicAdd(d + j, vals[j]);
    }
}

// ---------------------------------------------------------------------------
// Per-node finalize: +bias, +residual, LayerNorm, ReLU. One wave per node.
// ---------------------------------------------------------------------------
__global__ void node_finalize_kernel(const float* __restrict__ acc64,
                                     const float* __restrict__ xold,
                                     const float* __restrict__ bvec,
                                     const float* __restrict__ gamma,
                                     const float* __restrict__ beta,
                                     float* __restrict__ xnew) {
    const int lane = threadIdx.x & 31;
    const int n    = blockIdx.x * 8 + (threadIdx.x >> 5);
    if (n >= N_NODES) return;
    const float* a  = acc64 + (size_t)n * D_FEAT;
    const float* xo = xold + (size_t)n * D_FEAT;
    float v0 = a[lane]      + bvec[lane]      + xo[lane];
    float v1 = a[lane + 32] + bvec[lane + 32] + xo[lane + 32];

    float s = v0 + v1;
#pragma unroll
    for (int m = 1; m < 32; m <<= 1) s += __shfl_xor(s, m, 32);
    const float mu = s * (1.f / 64.f);

    const float d0 = v0 - mu, d1 = v1 - mu;
    float ss = d0 * d0 + d1 * d1;
#pragma unroll
    for (int m = 1; m < 32; m <<= 1) ss += __shfl_xor(ss, m, 32);
    const float rstd = rsqrtf(ss * (1.f / 64.f) + LN_EPS);

    float y0 = d0 * rstd * gamma[lane]      + beta[lane];
    float y1 = d1 * rstd * gamma[lane + 32] + beta[lane + 32];
    xnew[(size_t)n * D_FEAT + lane]      = y0 > 0.f ? y0 : 0.f;
    xnew[(size_t)n * D_FEAT + lane + 32] = y1 > 0.f ? y1 : 0.f;
}

// ---------------------------------------------------------------------------
// Per-layer scratch init.
// ---------------------------------------------------------------------------
__global__ void layer_init_kernel(float* __restrict__ acc64,
                                  float* __restrict__ nodemax,
                                  float* __restrict__ nodesum) {
    int i = blockIdx.x * 256 + threadIdx.x;
    if (i < N_NODES * D_FEAT) acc64[i] = 0.f;
    if (i < N_NODES * H_HEADS) {
        nodemax[i] = -__builtin_inff();
        nodesum[i] = 0.f;
    }
}

// ---------------------------------------------------------------------------
extern "C" void kernel_launch(void* const* d_in, const int* in_sizes, int n_in,
                              void* d_out, int out_size, void* d_ws, size_t ws_size,
                              hipStream_t stream) {
    const float* x   = (const float*)d_in[0];
    const float* Wl  = (const float*)d_in[1];
    const float* Wr  = (const float*)d_in[2];
    const float* att = (const float*)d_in[3];
    const float* b   = (const float*)d_in[4];
    const float* gm  = (const float*)d_in[5];
    const float* bt  = (const float*)d_in[6];
    const int*   ei  = (const int*)d_in[7];

    char* ws = (char*)d_ws;
    size_t off = 0;
    auto alloc = [&](size_t bytes) -> void* {
        void* p = ws + off;
        off = (off + bytes + 255) & ~(size_t)255;
        return p;
    };

    __bf16* wp   = (__bf16*)alloc((size_t)3 * 2 * 32 * 2 * 1024);      // 384 KB
    __bf16* xl   = (__bf16*)alloc((size_t)N_NODES * HD * 2);           // 51.2 MB
    __bf16* xr   = (__bf16*)alloc((size_t)N_NODES * HD * 2);           // 51.2 MB
    float*  elog = (float*)alloc((size_t)E_TOT * H_HEADS * 4);         // 14.4 MB
    float*  nmax = (float*)alloc((size_t)N_NODES * H_HEADS * 4);
    float*  nsum = (float*)alloc((size_t)N_NODES * H_HEADS * 4);
    float*  acc  = (float*)alloc((size_t)N_NODES * D_FEAT * 4);        // 12.8 MB
    float*  xb0  = (float*)alloc((size_t)N_NODES * D_FEAT * 4);
    float*  xb1  = (float*)alloc((size_t)N_NODES * D_FEAT * 4);

    pack_w_kernel<<<768, 256, 0, stream>>>(Wl, Wr, wp);

    const float* xin[3]  = {x, xb0, xb1};
    float*       xout[3] = {xb0, xb1, (float*)d_out};

    for (int l = 0; l < 3; ++l) {
        layer_init_kernel<<<(N_NODES * D_FEAT + 255) / 256, 256, 0, stream>>>(acc, nmax, nsum);
        gemm_xlxr_kernel<<<(N_NODES * 16) / (16 * 8) /* 6250 */, 256, 0, stream>>>(
            xin[l], wp, xl, xr, l);
        edge_logits_kernel<<<E_TOT / 8, 256, 0, stream>>>(ei, xl, xr, att + (size_t)l * HD,
                                                          elog, nmax);
        edge_expsum_kernel<<<(E_TOT * H_HEADS + 255) / 256, 256, 0, stream>>>(ei, elog,
                                                                              nmax, nsum);
        edge_aggregate_kernel<<<E_TOT / 8, 256, 0, stream>>>(ei, xl, elog, nsum, acc);
        node_finalize_kernel<<<N_NODES / 8, 256, 0, stream>>>(
            acc, xin[l], b + (size_t)l * D_FEAT, gm + (size_t)l * D_FEAT,
            bt + (size_t)l * D_FEAT, xout[l]);
    }
}
